// acLSTM_1511828488738
// MI455X (gfx1250) — matrix operand order repaired and checked by reading, not compiled, not run on hardware
//
#include <hip/hip_runtime.h>
#include <hip/hip_bf16.h>
#include <cstdint>
#include <cstddef>

#define IN_F   227
#define HID    1024
#define OUT_F  227
#define BATCH  256
#define SEQ    100
#define GATES  4096
#define KX1    256            // layer-1 input width padded 227 -> 256
#define HSZ    (BATCH * HID)  // one h/c state plane

#define KCHUNK  64            // K elements staged per LDS chunk
#define BROWS   256           // gate rows per block tile (4 quadrants x 64 units)
#define BSTRIDE 72            // halfs per staged row: 64 data + 8 pad (16B aligned)

typedef __attribute__((ext_vector_type(16))) __bf16 v16bf;
typedef __attribute__((ext_vector_type(8)))  float  v8f;
typedef __attribute__((ext_vector_type(4)))  unsigned int u32x4;

union FragU { v16bf v; u32x4 u[2]; };

// A fragment (16x32, bf16). Lane l<16: row M=l, K = k0..k0+15 contiguous.
// Lane l>=16: row M=l-16, K = k0+16..k0+31 contiguous (matches B fragments).
__device__ __forceinline__ v16bf load_a_frag(const __bf16* base, int ld, int row0,
                                             int k0, int lane) {
  const __bf16* p = base + (size_t)(row0 + (lane & 15)) * ld + k0 + ((lane >> 4) << 4);
  FragU f;
  f.u[0] = *(const u32x4*)(p);
  f.u[1] = *(const u32x4*)(p + 8);
  return f.v;
}

// B fragment (32x16, bf16) from an LDS-staged row-major slab [BROWS][BSTRIDE].
// Lane l<16: N=row0+l, K=ks..ks+15 contiguous. Lane l>=16: K=ks+16..ks+31.
__device__ __forceinline__ v16bf load_b_lds(const __bf16* sb, int row, int ks, int lane) {
  const __bf16* p = sb + (row + (lane & 15)) * BSTRIDE + ks + ((lane >> 4) << 4);
  FragU f;
  f.u[0] = *(const u32x4*)(p);      // -> ds_load_b128
  f.u[1] = *(const u32x4*)(p + 8);
  return f.v;
}

// B fragment directly from row-major global W[N,K] (used by small out-proj GEMM).
__device__ __forceinline__ v16bf load_b_frag(const __bf16* base, int ld, int nrow0,
                                             int k0, int lane) {
  const __bf16* p = base + (size_t)(nrow0 + (lane & 15)) * ld + k0 + ((lane >> 4) << 4);
  FragU f;
  f.u[0] = *(const u32x4*)(p);
  f.u[1] = *(const u32x4*)(p + 8);
  return f.v;
}

__device__ __forceinline__ v8f wmma_bf16(v16bf a, v16bf b, v8f c) {
  return __builtin_amdgcn_wmma_f32_16x16x32_bf16(false, a, false, b, (short)0, c,
                                                 false, false);
}

__device__ __forceinline__ float sigmoidf_(float x) {
  return 1.0f / (1.0f + __expf(-x));
}
__device__ __forceinline__ float tanhf_(float x) {
  float xc = fminf(fmaxf(x, -15.0f), 15.0f);
  float e = __expf(2.0f * xc);
  return (e - 1.0f) / (e + 1.0f);
}

// Async DMA of one 128-byte staged row (64 bf16) global -> LDS. The instruction's
// IOFFSET is added to BOTH the LDS and global addresses (ISA 10.7 async pseudocode),
// and both sides are contiguous within a row, so one base + immediate offsets works.
// Tracked by ASYNCcnt.
__device__ __forceinline__ void async_stage_row(const __bf16* g, const __bf16* l) {
  // Generic LDS address carries the LDS byte offset in addr[31:0].
  const unsigned loff = (unsigned)(uintptr_t)l;
  asm volatile("global_load_async_to_lds_b128 %0, %1, off"            :: "v"(loff), "v"(g) : "memory");
  asm volatile("global_load_async_to_lds_b128 %0, %1, off offset:16"  :: "v"(loff), "v"(g) : "memory");
  asm volatile("global_load_async_to_lds_b128 %0, %1, off offset:32"  :: "v"(loff), "v"(g) : "memory");
  asm volatile("global_load_async_to_lds_b128 %0, %1, off offset:48"  :: "v"(loff), "v"(g) : "memory");
  asm volatile("global_load_async_to_lds_b128 %0, %1, off offset:64"  :: "v"(loff), "v"(g) : "memory");
  asm volatile("global_load_async_to_lds_b128 %0, %1, off offset:80"  :: "v"(loff), "v"(g) : "memory");
  asm volatile("global_load_async_to_lds_b128 %0, %1, off offset:96"  :: "v"(loff), "v"(g) : "memory");
  asm volatile("global_load_async_to_lds_b128 %0, %1, off offset:112" :: "v"(loff), "v"(g) : "memory");
}

__device__ __forceinline__ void wait_asynccnt0() {
  asm volatile("s_wait_asynccnt 0" ::: "memory");
}

// ---------------------------------------------------------------------------
// Per-layer pointers for the fused 3-cell kernel (all layers are independent
// within a step because layers 2/3 consume the PRE-update h0/h1).
struct CellParams {
  const __bf16* X[3];    // layer input  [BATCH, kx] row-major bf16
  const __bf16* H[3];    // old hidden   [BATCH, HID]
  const __bf16* Wih[3];  // [GATES, kx]
  const __bf16* Whh[3];  // [GATES, HID]
  const float*  bih[3];
  const float*  bhh[3];
  float*        c[3];    // fp32 cell state, updated in place
  __bf16*       hout[3]; // new hidden (bf16)
  int           kx[3];   // 256 (layer1) or 1024
};

// Block = 256 threads (8 waves). Block tile: 64 batch rows x 64 hidden units,
// across all 4 gate quadrants (i,f,g,o at n, n+1024, n+2048, n+3072).
// Wave w: batch sub-tile (w>>2)*32, hidden sub-tile (w&3)*16.
// Weight slab for each 64-wide K chunk (256 gate rows x 64 K = 32 KB) is staged
// into LDS by async DMA, double buffered, so chunk c+1 streams in while chunk c
// feeds the WMMAs. Each weight byte is then read from L2 once per block.
__global__ __launch_bounds__(256, 1)
void cell_kernel(CellParams p) {
  __shared__ __attribute__((aligned(16))) __bf16 smem[2 * BROWS * BSTRIDE];

  const int layer = blockIdx.z;
  const int m0 = blockIdx.y * 64;
  const int j0 = blockIdx.x * 64;
  const int wave = threadIdx.x >> 5;
  const int lane = threadIdx.x & 31;
  const int mw = (wave >> 2) * 32;
  const int jw = (wave & 3) * 16;

  const __bf16* X   = p.X[layer];
  const __bf16* H   = p.H[layer];
  const __bf16* Wih = p.Wih[layer];
  const __bf16* Whh = p.Whh[layer];
  const int kx = p.kx[layer];

  const v8f vzero = {0.f, 0.f, 0.f, 0.f, 0.f, 0.f, 0.f, 0.f};
  v8f acc[4][2];
#pragma unroll
  for (int q = 0; q < 4; ++q) { acc[q][0] = vzero; acc[q][1] = vzero; }

  const int arow = m0 + mw;                 // + mi*16 for second tile
  const int ncol = j0 + jw;                 // hidden-unit base of this wave
  const int jj   = ncol + (lane & 15);      // this lane's hidden unit

  // Staged gate row owned by this thread: r = q*64 + local hidden unit.
  const int srow_q = threadIdx.x >> 6;      // quadrant 0..3
  const int srow_j = threadIdx.x & 63;      // hidden unit within tile

  const int c0 = kx >> 6;                   // chunks of the X@Wih^T segment
  const int nchunks = (kx + HID) >> 6;

  // --- stage chunk 0 ---
  {
    const __bf16* src = Wih + (size_t)((srow_q << 10) + j0 + srow_j) * kx;
    async_stage_row(src, &smem[threadIdx.x * BSTRIDE]);
  }

  for (int c = 0; c < nchunks; ++c) {
    const int buf = c & 1;
    wait_asynccnt0();       // this wave's DMA into buf has landed
    __syncthreads();        // everyone's DMA landed; everyone done reading buf^1

    if (c + 1 < nchunks) {  // kick DMA for the next chunk into the other buffer
      const int cn = c + 1;
      const __bf16* W; int ld, k0;
      if (cn < c0) { W = Wih; ld = kx;  k0 = cn << 6; }
      else         { W = Whh; ld = HID; k0 = (cn - c0) << 6; }
      const __bf16* src = W + (size_t)((srow_q << 10) + j0 + srow_j) * ld + k0;
      async_stage_row(src, &smem[(buf ^ 1) * (BROWS * BSTRIDE) + threadIdx.x * BSTRIDE]);
    }

    // compute on current chunk
    const __bf16* A; int lda, k0;
    if (c < c0) { A = X; lda = kx;  k0 = c << 6; }
    else        { A = H; lda = HID; k0 = (c - c0) << 6; }
    const __bf16* sb = &smem[buf * (BROWS * BSTRIDE)];
#pragma unroll
    for (int s = 0; s < 2; ++s) {
      const int ks = s << 5;
      // Materialize ALL fragments first (independent dest regs) so the
      // scheduler can clause the ds_loads and overlap waits with WMMAs,
      // instead of the load->wait0->wmma serial chain.
      v16bf a0 = load_a_frag(A, lda, arow,      k0 + ks, lane);
      v16bf a1 = load_a_frag(A, lda, arow + 16, k0 + ks, lane);
      v16bf b0 = load_b_lds(sb, jw,       ks, lane);
      v16bf b1 = load_b_lds(sb, 64 + jw,  ks, lane);
      v16bf b2 = load_b_lds(sb, 128 + jw, ks, lane);
      v16bf b3 = load_b_lds(sb, 192 + jw, ks, lane);
      acc[0][0] = wmma_bf16(a0, b0, acc[0][0]);
      acc[0][1] = wmma_bf16(a1, b0, acc[0][1]);
      acc[1][0] = wmma_bf16(a0, b1, acc[1][0]);
      acc[1][1] = wmma_bf16(a1, b1, acc[1][1]);
      acc[2][0] = wmma_bf16(a0, b2, acc[2][0]);
      acc[2][1] = wmma_bf16(a1, b2, acc[2][1]);
      acc[3][0] = wmma_bf16(a0, b3, acc[3][0]);
      acc[3][1] = wmma_bf16(a1, b3, acc[3][1]);
    }
  }

  // Fused LSTM epilogue. C-layout: element (vgpr r, lane l) is
  // M = r + 8*(l>>4), N = l&15 -> all four gates of one unit land in the same
  // lane at the same vgpr index across the quadrant accumulators.
  const float* bih = p.bih[layer];
  const float* bhh = p.bhh[layer];
  float*   cptr = p.c[layer];
  __bf16*  hout = p.hout[layer];

  const float b_i = bih[jj]           + bhh[jj];
  const float b_f = bih[HID + jj]     + bhh[HID + jj];
  const float b_g = bih[2 * HID + jj] + bhh[2 * HID + jj];
  const float b_o = bih[3 * HID + jj] + bhh[3 * HID + jj];

#pragma unroll
  for (int mi = 0; mi < 2; ++mi) {
#pragma unroll
    for (int r = 0; r < 8; ++r) {
      const int m = m0 + mw + mi * 16 + r + ((lane >> 4) << 3);
      const size_t idx = (size_t)m * HID + jj;
      const float iv = sigmoidf_(acc[0][mi][r] + b_i);
      const float fv = sigmoidf_(acc[1][mi][r] + b_f);
      const float gv = tanhf_(acc[2][mi][r] + b_g);
      const float ov = sigmoidf_(acc[3][mi][r] + b_o);
      const float cn = fv * cptr[idx] + iv * gv;
      cptr[idx] = cn;
      hout[idx] = (__bf16)(ov * tanhf_(cn));
    }
  }
}

// out[:, t] = h2_new @ Wd^T + bd   (Wd padded to 256 rows in bf16)
__global__ __launch_bounds__(256, 1)
void out_kernel(const __bf16* __restrict__ h2, const __bf16* __restrict__ wd,
                const float* __restrict__ bd, float* __restrict__ out, int t) {
  const int m0 = blockIdx.y * 64;
  const int n0 = blockIdx.x * 64;
  const int wave = threadIdx.x >> 5;
  const int lane = threadIdx.x & 31;
  const int mw = (wave >> 2) * 32;
  const int nw = (wave & 3) * 16;

  const v8f vzero = {0.f, 0.f, 0.f, 0.f, 0.f, 0.f, 0.f, 0.f};
  v8f acc0 = vzero, acc1 = vzero;
  const int arow = m0 + mw;
  const int ncol = n0 + nw;

  for (int k0 = 0; k0 < HID; k0 += 32) {
    v16bf a0 = load_a_frag(h2, HID, arow,      k0, lane);
    v16bf a1 = load_a_frag(h2, HID, arow + 16, k0, lane);
    v16bf b  = load_b_frag(wd, HID, ncol, k0, lane);
    acc0 = wmma_bf16(a0, b, acc0);
    acc1 = wmma_bf16(a1, b, acc1);
  }

  const int nn = ncol + (lane & 15);
  if (nn < OUT_F) {
    const float bias = bd[nn];
#pragma unroll
    for (int r = 0; r < 8; ++r) {
      int m = m0 + mw + r + ((lane >> 4) << 3);
      out[(size_t)m * (SEQ * OUT_F) + (size_t)t * OUT_F + nn] = acc0[r] + bias;
      m += 16;
      out[(size_t)m * (SEQ * OUT_F) + (size_t)t * OUT_F + nn] = acc1[r] + bias;
    }
  }
}

// Build layer-1 input frame: teacher-forced gt frame or previous output,
// padded 227 -> 256 and rounded to bf16.
__global__ void prep_x_kernel(const float* __restrict__ seq,
                              const float* __restrict__ out,
                              __bf16* __restrict__ xin, int t, int use_gt) {
  const int b = blockIdx.x;
  const int f = threadIdx.x;
  float v = 0.0f;
  if (f < IN_F) {
    v = use_gt ? seq[(size_t)b * (SEQ * IN_F) + (size_t)t * IN_F + f]
               : out[(size_t)b * (SEQ * OUT_F) + (size_t)(t - 1) * OUT_F + f];
  }
  xin[b * KX1 + f] = (__bf16)v;
}

// fp32 -> bf16 weight conversion with zero padding.
__global__ void conv_pad_kernel(const float* __restrict__ src, __bf16* __restrict__ dst,
                                int srows, int scols, int drows, int dcols) {
  const long total = (long)drows * dcols;
  const long stride = (long)gridDim.x * blockDim.x;
  for (long i = (long)blockIdx.x * blockDim.x + threadIdx.x; i < total; i += stride) {
    const int r = (int)(i / dcols);
    const int c = (int)(i % dcols);
    float v = (r < srows && c < scols) ? src[(long)r * scols + c] : 0.0f;
    dst[i] = (__bf16)v;
  }
}

// ---------------------------------------------------------------------------
extern "C" void kernel_launch(void* const* d_in, const int* in_sizes, int n_in,
                              void* d_out, int out_size, void* d_ws, size_t ws_size,
                              hipStream_t stream) {
  (void)in_sizes; (void)n_in; (void)out_size; (void)ws_size;

  const float* seq  = (const float*)d_in[0];
  const float* Wih1 = (const float*)d_in[1];
  const float* Whh1 = (const float*)d_in[2];
  const float* bih1 = (const float*)d_in[3];
  const float* bhh1 = (const float*)d_in[4];
  const float* Wih2 = (const float*)d_in[5];
  const float* Whh2 = (const float*)d_in[6];
  const float* bih2 = (const float*)d_in[7];
  const float* bhh2 = (const float*)d_in[8];
  const float* Wih3 = (const float*)d_in[9];
  const float* Whh3 = (const float*)d_in[10];
  const float* bih3 = (const float*)d_in[11];
  const float* bhh3 = (const float*)d_in[12];
  const float* Wd   = (const float*)d_in[13];
  const float* bd   = (const float*)d_in[14];
  float* out = (float*)d_out;

  // Workspace carve-out (256B aligned regions).
  char* ws = (char*)d_ws;
  size_t off = 0;
  auto take = [&](size_t bytes) -> void* {
    void* p = ws + off;
    off += (bytes + 255) & ~(size_t)255;
    return p;
  };
  __bf16* wih1p = (__bf16*)take((size_t)GATES * KX1 * 2);
  __bf16* whh1  = (__bf16*)take((size_t)GATES * HID * 2);
  __bf16* wih2  = (__bf16*)take((size_t)GATES * HID * 2);
  __bf16* whh2  = (__bf16*)take((size_t)GATES * HID * 2);
  __bf16* wih3  = (__bf16*)take((size_t)GATES * HID * 2);
  __bf16* whh3  = (__bf16*)take((size_t)GATES * HID * 2);
  __bf16* wdp   = (__bf16*)take((size_t)256 * HID * 2);
  __bf16* xin   = (__bf16*)take((size_t)BATCH * KX1 * 2);
  __bf16* hbuf0 = (__bf16*)take((size_t)3 * HSZ * 2);
  __bf16* hbuf1 = (__bf16*)take((size_t)3 * HSZ * 2);
  float*  cbuf  = (float*)take((size_t)3 * HSZ * 4);

  // One-time (per launch) weight conversion + state zeroing.
  conv_pad_kernel<<<1024, 256, 0, stream>>>(Wih1, wih1p, GATES, IN_F, GATES, KX1);
  conv_pad_kernel<<<1024, 256, 0, stream>>>(Whh1, whh1, GATES, HID, GATES, HID);
  conv_pad_kernel<<<1024, 256, 0, stream>>>(Wih2, wih2, GATES, HID, GATES, HID);
  conv_pad_kernel<<<1024, 256, 0, stream>>>(Whh2, whh2, GATES, HID, GATES, HID);
  conv_pad_kernel<<<1024, 256, 0, stream>>>(Wih3, wih3, GATES, HID, GATES, HID);
  conv_pad_kernel<<<1024, 256, 0, stream>>>(Whh3, whh3, GATES, HID, GATES, HID);
  conv_pad_kernel<<<256, 256, 0, stream>>>(Wd, wdp, OUT_F, HID, 256, HID);
  hipMemsetAsync(hbuf0, 0, (size_t)3 * HSZ * 2, stream);
  hipMemsetAsync(hbuf1, 0, (size_t)3 * HSZ * 2, stream);
  hipMemsetAsync(cbuf, 0, (size_t)3 * HSZ * 4, stream);

  const dim3 cgrid(HID / 64, BATCH / 64, 3);  // (hidden tiles, batch tiles, layer)
  const dim3 ogrid(4, BATCH / 64);            // 256 (padded) out cols, batch tiles

  for (int t = 0; t < SEQ; ++t) {
    const int use_gt = ((t % 10) < 5) ? 1 : 0;  // 5 gt frames, 5 conditioned
    prep_x_kernel<<<BATCH, 256, 0, stream>>>(seq, out, xin, t, use_gt);

    __bf16* hold = (t & 1) ? hbuf1 : hbuf0;
    __bf16* hnew = (t & 1) ? hbuf0 : hbuf1;

    CellParams p;
    p.X[0] = xin;            p.kx[0] = KX1;
    p.X[1] = hold + 0 * HSZ; p.kx[1] = HID;   // old h0 (pre-update) -> layer 2
    p.X[2] = hold + 1 * HSZ; p.kx[2] = HID;   // old h1 (pre-update) -> layer 3
    p.H[0] = hold + 0 * HSZ;
    p.H[1] = hold + 1 * HSZ;
    p.H[2] = hold + 2 * HSZ;
    p.Wih[0] = wih1p; p.Wih[1] = wih2; p.Wih[2] = wih3;
    p.Whh[0] = whh1;  p.Whh[1] = whh2; p.Whh[2] = whh3;
    p.bih[0] = bih1; p.bih[1] = bih2; p.bih[2] = bih3;
    p.bhh[0] = bhh1; p.bhh[1] = bhh2; p.bhh[2] = bhh3;
    p.c[0] = cbuf + 0 * HSZ; p.c[1] = cbuf + 1 * HSZ; p.c[2] = cbuf + 2 * HSZ;
    p.hout[0] = hnew + 0 * HSZ;
    p.hout[1] = hnew + 1 * HSZ;
    p.hout[2] = hnew + 2 * HSZ;

    cell_kernel<<<cgrid, 256, 0, stream>>>(p);
    out_kernel<<<ogrid, 256, 0, stream>>>(hnew + 2 * HSZ, wdp, bd, out, t);
  }
}
